// Attention_20418274525510
// MI455X (gfx1250) — matrix operand order repaired and checked by reading, not compiled, not run on hardware
//
#include <hip/hip_runtime.h>

// ---------------- problem constants ----------------
#define DIMC   256
#define WSZ    8
#define NPIX   64          // 8x8 window
#define NHEADS 8
#define DHEAD  32
#define D4     64
#define EPS    1e-5f
#define ATT_SCALE 0.17677669529663687f   // 32^-0.5

// LDS row strides (elements) — all 16B multiples for b128 access, odd bank groups
#define XR 264             // rows of 256 (f32 or f16)
#define QS 40              // rows of 32 f16 (Q/K transposed: [pix][d])
#define VS 72              // rows of 64 f16 (V: [d][pix], attn: [i][j])

typedef __attribute__((ext_vector_type(16))) _Float16 v16h;
typedef __attribute__((ext_vector_type(8)))  _Float16 v8h;
typedef __attribute__((ext_vector_type(8)))  float    v8f;

__device__ __forceinline__ v16h concat8(v8h lo, v8h hi) {
  return __builtin_shufflevector(lo, hi, 0, 1, 2, 3, 4, 5, 6, 7,
                                 8, 9, 10, 11, 12, 13, 14, 15);
}

// WMMA 16x16x32 f16 per-lane layouts used by this code (contiguous k, A and B alike):
//  A (16x32): M = lane%16; K run:  [offa, offa+16), offa = (lane>=16)?16:0
//  B (32x16): N = lane%16; K run:  [offb, offb+16), offb = (lane>=16)?16:0
//  C (f32)  : reg r -> M = r + ((lane>=16)?8:0), N = lane%16

// ---------------- workspace (d_ws) layout (bytes) ----------------
constexpr size_t WS_BIAS = 0;                         // 289 f32
constexpr size_t WS_QKVH = 4096;                      // f16[768*256] = 393216
constexpr size_t WS_OUTH = WS_QKVH + 393216;          // f16[256*256] = 131072

// ---------------- LDS layout (bytes) ----------------
constexpr int OFF_BIAS = 0;                    // 289 f32 (pad 1216)
constexpr int OFF_G    = 1216;                 // 256 f32
constexpr int OFF_B    = 2240;                 // 256 f32
constexpr int OFF_SUM  = 3264;                 // 4*64 f32
constexpr int OFF_SQ   = 4288;                 // 4*64 f32
constexpr int OFF_MRS  = 5312;                 // 64*2 f32
constexpr int OFF_OVL  = 5824;                 // overlay: xrawT f32[64][264] = 67584
                                               //          attn  f16[8][64][72] = 73728
constexpr int OFF_XLN  = OFF_OVL + 73728;      // f16[64][264] = 33792  (x^T, LN'ed)
constexpr int OFF_QT   = OFF_XLN + 33792;      // f16[8][64][40] = 40960 (Q^T, pre-scaled)
constexpr int OFF_KT   = OFF_QT  + 40960;      // f16[8][64][40] = 40960 (K^T per head)
constexpr int OFF_V    = OFF_KT  + 40960;      // f16[8][32][72] = 36864 (V per head)
constexpr int OFF_OUT  = OFF_V   + 36864;      // f16[64][264] = 33792  (attn-out ^T)
constexpr int LDS_TOTAL= OFF_OUT + 33792;      // 265920 B  (< 320 KB)

// ================= one-shot weight f32 -> f16 conversion ===================
__global__ void weight_cvt_kernel(const float* __restrict__ w_qkv,
                                  const float* __restrict__ w_out,
                                  _Float16* __restrict__ qkv_h,
                                  _Float16* __restrict__ out_h) {
  const int t = blockIdx.x * blockDim.x + threadIdx.x;   // 65536 threads
  for (int g = t; g < (768 * 256) / 4; g += 65536) {
    float4 v = *(const float4*)(w_qkv + g * 4);
    qkv_h[g * 4 + 0] = (_Float16)v.x; qkv_h[g * 4 + 1] = (_Float16)v.y;
    qkv_h[g * 4 + 2] = (_Float16)v.z; qkv_h[g * 4 + 3] = (_Float16)v.w;
  }
  for (int g = t; g < (256 * 256) / 4; g += 65536) {
    float4 v = *(const float4*)(w_out + g * 4);
    out_h[g * 4 + 0] = (_Float16)v.x; out_h[g * 4 + 1] = (_Float16)v.y;
    out_h[g * 4 + 2] = (_Float16)v.z; out_h[g * 4 + 3] = (_Float16)v.w;
  }
}

// ================= dynamic position bias MLP (tiny; run once) ==============
__device__ __forceinline__ void ln_relu64(float* h, const float* __restrict__ g,
                                          const float* __restrict__ be) {
  float m = 0.f;
  for (int j = 0; j < D4; ++j) m += h[j];
  m *= (1.f / D4);
  float v = 0.f;
  for (int j = 0; j < D4; ++j) { float d = h[j] - m; v += d * d; }
  v *= (1.f / D4);
  float r = rsqrtf(v + EPS);
  for (int j = 0; j < D4; ++j) {
    float y = (h[j] - m) * r * g[j] + be[j];
    h[j] = y > 0.f ? y : 0.f;
  }
}

__global__ void dpb_kernel(const float* __restrict__ w1, const float* __restrict__ b1,
                           const float* __restrict__ g1, const float* __restrict__ be1,
                           const float* __restrict__ w2, const float* __restrict__ b2,
                           const float* __restrict__ g2, const float* __restrict__ be2,
                           const float* __restrict__ w3, const float* __restrict__ b3,
                           const float* __restrict__ g3, const float* __restrict__ be3,
                           const float* __restrict__ w4, const float* __restrict__ b4,
                           float* __restrict__ biasOut) {
  int t = threadIdx.x;
  if (t >= 289) return;
  float py = (float)(t / 17) - 8.0f;
  float px = (float)(t % 17) - 8.0f;
  float h[D4], h2[D4];
  for (int j = 0; j < D4; ++j) h[j] = py * w1[2 * j] + px * w1[2 * j + 1] + b1[j];
  ln_relu64(h, g1, be1);
  for (int j = 0; j < D4; ++j) {
    float s = b2[j];
    for (int k = 0; k < D4; ++k) s += w2[j * D4 + k] * h[k];
    h2[j] = s;
  }
  ln_relu64(h2, g2, be2);
  for (int j = 0; j < D4; ++j) {
    float s = b3[j];
    for (int k = 0; k < D4; ++k) s += w3[j * D4 + k] * h2[k];
    h[j] = s;
  }
  ln_relu64(h, g3, be3);
  float o = b4[0];
  for (int k = 0; k < D4; ++k) o += w4[k] * h[k];
  biasOut[t] = o;
}

// ================= fused windowed attention: 1 workgroup per window ========
__global__ __launch_bounds__(256)
void win_attn_kernel(const float* __restrict__ x,
                     const float* __restrict__ norm_g,
                     const float* __restrict__ norm_b,
                     const _Float16* __restrict__ w_qkv_h,
                     const _Float16* __restrict__ w_out_h,
                     const float* __restrict__ b_out,
                     const float* __restrict__ biasTab,
                     float* __restrict__ out) {
  extern __shared__ char smem[];
  float*    sBias  = (float*)(smem + OFF_BIAS);
  float*    sG     = (float*)(smem + OFF_G);
  float*    sB     = (float*)(smem + OFF_B);
  float*    sSum   = (float*)(smem + OFF_SUM);
  float*    sSq    = (float*)(smem + OFF_SQ);
  float*    sMRS   = (float*)(smem + OFF_MRS);
  float*    sXrawT = (float*)(smem + OFF_OVL);      // [pix][ch], dead after LN
  _Float16* sAttn  = (_Float16*)(smem + OFF_OVL);   // [head][i][j] overlay
  _Float16* sXlnT  = (_Float16*)(smem + OFF_XLN);   // [pix][ch]
  _Float16* sQT    = (_Float16*)(smem + OFF_QT);    // [head][pix][d] (pre-scaled)
  _Float16* sKT    = (_Float16*)(smem + OFF_KT);    // [head][pix][d]
  _Float16* sV     = (_Float16*)(smem + OFF_V);     // [head][d][pix]
  _Float16* sOutT  = (_Float16*)(smem + OFF_OUT);   // [pix][ch]

  const int tid  = threadIdx.x;
  const int lane = tid & 31;
  const int wv   = tid >> 5;
  const int nlo  = lane & 15;
  const int hi8  = (lane & 16) ? 8 : 0;     // C-row offset
  const int offa = (lane & 16) ? 16 : 0;    // A K-run base
  const int offb = (lane & 16) ? 16 : 0;    // B K-run base

  const int widx = blockIdx.x;              // b*1024 + wh*32 + ww
  const int bb   = widx >> 10;
  const int rem  = widx & 1023;
  const int h0   = (rem >> 5) * WSZ;
  const int w0   = (rem & 31) * WSZ;
  const size_t xbase = (size_t)bb * DIMC * 65536;

  // ---- stage small tables ----
  for (int i = tid; i < 289; i += 256) sBias[i] = biasTab[i];
  for (int i = tid; i < DIMC; i += 256) { sG[i] = norm_g[i]; sB[i] = norm_b[i]; }

  // ---- load window: float4 global loads, transpose into [pix][ch] ----
  for (int i = tid; i < DIMC * 16; i += 256) {
    int ch = i >> 4;
    int p4 = (i & 15) * 4;                  // 4 pixels within one window row
    const float4 v = *(const float4*)(x + xbase + (size_t)ch * 65536 +
                                      (size_t)(h0 + (p4 >> 3)) * 256 + w0 + (p4 & 7));
    sXrawT[(p4 + 0) * XR + ch] = v.x;
    sXrawT[(p4 + 1) * XR + ch] = v.y;
    sXrawT[(p4 + 2) * XR + ch] = v.z;
    sXrawT[(p4 + 3) * XR + ch] = v.w;
  }
  __syncthreads();

  // ---- channel-first LayerNorm (stats over 256 ch per pixel) ----
  const int pP = tid & 63, qQ = tid >> 6;
  {
    const float* rowp = sXrawT + pP * XR + qQ * 64;
    float s = 0.f, sq = 0.f;
    for (int c = 0; c < 64; c += 4) {
      float4 v = *(const float4*)(rowp + c);
      s += v.x + v.y + v.z + v.w;
      sq += v.x * v.x + v.y * v.y + v.z * v.z + v.w * v.w;
    }
    sSum[qQ * 64 + pP] = s; sSq[qQ * 64 + pP] = sq;
  }
  __syncthreads();
  if (tid < 64) {
    float s  = sSum[tid] + sSum[64 + tid] + sSum[128 + tid] + sSum[192 + tid];
    float sq = sSq[tid]  + sSq[64 + tid]  + sSq[128 + tid]  + sSq[192 + tid];
    float m  = s * (1.f / 256.f);
    float v  = sq * (1.f / 256.f) - m * m;
    sMRS[tid * 2]     = m;
    sMRS[tid * 2 + 1] = rsqrtf(v + EPS);
  }
  __syncthreads();
  {
    const float* rowp = sXrawT + pP * XR + qQ * 64;
    float m = sMRS[pP * 2], rs = sMRS[pP * 2 + 1];
    for (int c0 = 0; c0 < 64; c0 += 8) {
      v8h hv;
#pragma unroll
      for (int j = 0; j < 8; ++j) {
        int ch = qQ * 64 + c0 + j;
        hv[j] = (_Float16)((rowp[c0 + j] - m) * rs * sG[ch] + sB[ch]);
      }
      *(v8h*)(sXlnT + pP * XR + qQ * 64 + c0) = hv;
    }
  }
  __syncthreads();

  // ---- QKV GEMM: wave computes 96 rows of (768x256)@(256x64) ----
  for (int mt = 0; mt < 6; ++mt) {
    const int rbase = wv * 96 + mt * 16;      // multiple of 16; never crosses Q/K/V
    const int mrow  = rbase + nlo;
    v16h areg[8];
#pragma unroll
    for (int kc = 0; kc < 8; ++kc) {
      const _Float16* wr = w_qkv_h + mrow * 256 + kc * 32 + offa;
      areg[kc] = concat8(*(const v8h*)wr, *(const v8h*)(wr + 8));
    }
    const int ty    = rbase >> 8;             // 0=Q 1=K 2=V
    const int head  = (rbase >> 5) & 7;
    const int dbase = (rbase & 31) + hi8;
    const float cscale = (ty == 0) ? ATT_SCALE : 1.0f;   // fold q scale here
#pragma unroll
    for (int nt = 0; nt < 4; ++nt) {
      v8f acc = {};
      const _Float16* bp = sXlnT + (nt * 16 + nlo) * XR + offb;
#pragma unroll
      for (int kc = 0; kc < 8; ++kc) {
        v16h b = concat8(*(const v8h*)(bp + kc * 32),
                         *(const v8h*)(bp + kc * 32 + 8));
        acc = __builtin_amdgcn_wmma_f32_16x16x32_f16(false, areg[kc], false, b,
                                                     (short)0, acc, false, false);
      }
      const int pix = nt * 16 + nlo;
      if (ty < 2) {                           // Q/K -> [head][pix][d], packed b128
        v8h hv;
#pragma unroll
        for (int r = 0; r < 8; ++r) hv[r] = (_Float16)(acc[r] * cscale);
        _Float16* dst = (ty == 0 ? sQT : sKT) + (head * 64 + pix) * QS + dbase;
        *(v8h*)dst = hv;
      } else {                                // V -> [head][d][pix], true transpose
#pragma unroll
        for (int r = 0; r < 8; ++r)
          sV[(head * 32 + dbase + r) * VS + pix] = (_Float16)acc[r];
      }
    }
  }
  __syncthreads();

  // ---- attention: wave wv == head wv ----
  const _Float16* q0  = sQT + wv * 64 * QS;
  const _Float16* k0  = sKT + wv * 64 * QS;
  const _Float16* v0  = sV  + wv * 32 * VS;
  _Float16*       at0 = sAttn + wv * 64 * VS;   // [i][j]

  // preload all K B-tiles (B: N=j-col, K=d run of 16)
  v16h bk[4];
#pragma unroll
  for (int jt = 0; jt < 4; ++jt) {
    const _Float16* kp = k0 + (jt * 16 + nlo) * QS + offb;
    bk[jt] = concat8(*(const v8h*)kp, *(const v8h*)(kp + 8));
  }
  for (int it = 0; it < 4; ++it) {
    const _Float16* qp = q0 + (it * 16 + nlo) * QS + offa;
    v16h aq = concat8(*(const v8h*)qp, *(const v8h*)(qp + 8));
    v8f acc[4];
#pragma unroll
    for (int jt = 0; jt < 4; ++jt) {
      v8f z = {};
      acc[jt] = __builtin_amdgcn_wmma_f32_16x16x32_f16(false, aq, false, bk[jt],
                                                       (short)0, z, false, false);
    }
    // rel-pos bias + row softmax (rows split across 16-lane halves)
#pragma unroll
    for (int r = 0; r < 8; ++r) {
      int i  = it * 16 + r + hi8;
      int iy = i >> 3, ix = i & 7;
      float mrow = -3.4e38f;
      float sv[4];
#pragma unroll
      for (int jt = 0; jt < 4; ++jt) {
        int j  = jt * 16 + nlo;
        int dy = iy - (j >> 3) + 7, dx = ix - (j & 7) + 7;
        float s = acc[jt][r] + sBias[dy * 15 + dx];
        sv[jt] = s;
        mrow = fmaxf(mrow, s);
      }
      for (int off = 8; off > 0; off >>= 1)
        mrow = fmaxf(mrow, __shfl_xor(mrow, off, 32));
      float sum = 0.f;
#pragma unroll
      for (int jt = 0; jt < 4; ++jt) { sv[jt] = __expf(sv[jt] - mrow); sum += sv[jt]; }
      for (int off = 8; off > 0; off >>= 1)
        sum += __shfl_xor(sum, off, 32);
      float inv = 1.f / sum;
#pragma unroll
      for (int jt = 0; jt < 4; ++jt)
        at0[i * VS + jt * 16 + nlo] = (_Float16)(sv[jt] * inv);
    }
  }

  // ---- out^T = V (32x64) @ attn^T (64x64); same-wave LDS ops in order ----
  v16h av[2][2];
#pragma unroll
  for (int mtd = 0; mtd < 2; ++mtd)
#pragma unroll
    for (int kj = 0; kj < 2; ++kj) {
      const _Float16* vp = v0 + (mtd * 16 + nlo) * VS + kj * 32 + offa;
      av[mtd][kj] = concat8(*(const v8h*)vp, *(const v8h*)(vp + 8));
    }
  for (int nti = 0; nti < 4; ++nti) {
    const _Float16* ap = at0 + (nti * 16 + nlo) * VS;   // B: N=i, K=j run
    v16h bt[2];
#pragma unroll
    for (int kj = 0; kj < 2; ++kj)
      bt[kj] = concat8(*(const v8h*)(ap + kj * 32 + offb),
                       *(const v8h*)(ap + kj * 32 + offb + 8));
#pragma unroll
    for (int mtd = 0; mtd < 2; ++mtd) {
      v8f acc = {};
#pragma unroll
      for (int kj = 0; kj < 2; ++kj)
        acc = __builtin_amdgcn_wmma_f32_16x16x32_f16(false, av[mtd][kj], false, bt[kj],
                                                     (short)0, acc, false, false);
      // C: M = d = mtd*16 + r + hi8, N = i; store [pix][ch] packed b128
      v8h hv;
#pragma unroll
      for (int r = 0; r < 8; ++r) hv[r] = (_Float16)acc[r];
      *(v8h*)(sOutT + (nti * 16 + nlo) * XR + wv * 32 + mtd * 16 + hi8) = hv;
    }
  }
  __syncthreads();

  // ---- final projection: y = w_out(256x256) @ out(256x64) + b_out ----
  for (int mt = 0; mt < 2; ++mt) {
    const int rbase = wv * 32 + mt * 16;
    const int mrow  = rbase + nlo;
    v16h areg[8];
#pragma unroll
    for (int kc = 0; kc < 8; ++kc) {
      const _Float16* wr = w_out_h + mrow * 256 + kc * 32 + offa;
      areg[kc] = concat8(*(const v8h*)wr, *(const v8h*)(wr + 8));
    }
    float bo[8];
#pragma unroll
    for (int r = 0; r < 8; ++r) bo[r] = b_out[rbase + r + hi8];
#pragma unroll
    for (int nt = 0; nt < 4; ++nt) {
      v8f acc = {};
      const _Float16* bp = sOutT + (nt * 16 + nlo) * XR + offb;
#pragma unroll
      for (int kc = 0; kc < 8; ++kc) {
        v16h b = concat8(*(const v8h*)(bp + kc * 32),
                         *(const v8h*)(bp + kc * 32 + 8));
        acc = __builtin_amdgcn_wmma_f32_16x16x32_f16(false, areg[kc], false, b,
                                                     (short)0, acc, false, false);
      }
      const int p = nt * 16 + nlo;
      const size_t obase =
          (size_t)bb * DIMC * 65536 + (size_t)(h0 + (p >> 3)) * 256 + w0 + (p & 7);
#pragma unroll
      for (int r = 0; r < 8; ++r)
        out[obase + (size_t)(rbase + r + hi8) * 65536] = acc[r] + bo[r];
    }
  }
}

// =========================== host launcher =================================
extern "C" void kernel_launch(void* const* d_in, const int* in_sizes, int n_in,
                              void* d_out, int out_size, void* d_ws, size_t ws_size,
                              hipStream_t stream) {
  (void)in_sizes; (void)n_in; (void)out_size; (void)ws_size;
  const float* x      = (const float*)d_in[0];
  const float* norm_g = (const float*)d_in[1];
  const float* norm_b = (const float*)d_in[2];
  const float* w_qkv  = (const float*)d_in[3];
  const float* w_out  = (const float*)d_in[4];
  const float* b_out  = (const float*)d_in[5];
  const float* w1  = (const float*)d_in[6];
  const float* b1  = (const float*)d_in[7];
  const float* g1  = (const float*)d_in[8];
  const float* be1 = (const float*)d_in[9];
  const float* w2  = (const float*)d_in[10];
  const float* b2  = (const float*)d_in[11];
  const float* g2  = (const float*)d_in[12];
  const float* be2 = (const float*)d_in[13];
  const float* w3  = (const float*)d_in[14];
  const float* b3  = (const float*)d_in[15];
  const float* g3  = (const float*)d_in[16];
  const float* be3 = (const float*)d_in[17];
  const float* w4  = (const float*)d_in[18];
  const float* b4  = (const float*)d_in[19];

  float*    biasTab = (float*)((char*)d_ws + WS_BIAS);   // 289 f32
  _Float16* qkv_h   = (_Float16*)((char*)d_ws + WS_QKVH);
  _Float16* out_h   = (_Float16*)((char*)d_ws + WS_OUTH);
  float*    out     = (float*)d_out;

  dpb_kernel<<<1, 320, 0, stream>>>(w1, b1, g1, be1, w2, b2, g2, be2,
                                    w3, b3, g3, be3, w4, b4, biasTab);
  weight_cvt_kernel<<<256, 256, 0, stream>>>(w_qkv, w_out, qkv_h, out_h);
  win_attn_kernel<<<2048, 256, LDS_TOTAL, stream>>>(
      x, norm_g, norm_b, qkv_h, out_h, b_out, biasTab, out);
}
